// CompositionNet_87771951661334
// MI455X (gfx1250) — matrix operand order repaired and checked by reading, not compiled, not run on hardware
//
#include <hip/hip_runtime.h>
#include <hip/hip_bf16.h>
#include <math.h>

#define NATOMS 50000
#define MNBR   12
#define NCRYS  5000
#define ORIGF  200
#define NBRF   64
#define AFL    64
#define NG     3
#define HF     128

typedef __attribute__((ext_vector_type(16))) __bf16 v16bf;
typedef __attribute__((ext_vector_type(8)))  float  v8f;

union BF16x16 { v16bf v; unsigned short s[16]; uint4 q[2]; };
union BF8PK   { unsigned short s[8]; uint4 q; };

__device__ inline unsigned short f2bf(float f) {
    unsigned int u = __float_as_uint(f);
    unsigned int r = u + 0x7FFFu + ((u >> 16) & 1u);   // round-to-nearest-even
    return (unsigned short)(r >> 16);
}
__device__ inline float softplusf(float x) {
    return fmaxf(x, 0.0f) + log1pf(expf(-fabsf(x)));
}
__device__ inline float sigmoidf(float x) {
    return 1.0f / (1.0f + expf(-x));
}

// Build a 16x32 bf16 A fragment from an LDS-staged row-major *bf16* tile.
// CDNA5 layout: lanes 0-15 hold row m, K = kbase+[0..7] (halves 0-7) and
// kbase+[16..23] (halves 8-15); lanes 16-31 hold K offsets +8 / +24.
// Pure data movement: two 16B LDS loads, zero conversion VALU.
__device__ inline v16bf load_a_frag_bf(const unsigned short* A, int lane, int kbase, int stride) {
    int m  = lane & 15;
    int kh = (lane >> 4) << 3;      // 0 or 8
    const unsigned short* p = A + m * stride + kbase + kh;
    BF16x16 r;
    r.q[0] = *(const uint4*)p;          // K = kbase+kh .. +7
    r.q[1] = *(const uint4*)(p + 16);   // K = kbase+16+kh .. +7
    return r.v;
}

// Bulk f32 -> bf16 conversion, 8 elements/thread (bandwidth-bound).
__global__ __launch_bounds__(256) void conv_bf16(const float* __restrict__ src,
                                                 unsigned short* __restrict__ dst, int n8) {
    int i = blockIdx.x * 256 + threadIdx.x;
    if (i >= n8) return;
    float4 a = *((const float4*)src + (size_t)i * 2);
    float4 b = *((const float4*)src + (size_t)i * 2 + 1);
    BF8PK r;
    r.s[0] = f2bf(a.x); r.s[1] = f2bf(a.y); r.s[2] = f2bf(a.z); r.s[3] = f2bf(a.w);
    r.s[4] = f2bf(b.x); r.s[5] = f2bf(b.y); r.s[6] = f2bf(b.z); r.s[7] = f2bf(b.w);
    *((uint4*)dst + i) = r.q;
}

// Pack a row-major (K x N) f32 weight into WMMA B-fragment order (bf16):
// tile (kt,nt): 32 lanes x 16 halves contiguous. B(k,n): lane = (n%16)+16*(((k%32)/8)%2),
// half = (k%8)+8*((k%32)/16) within the 32-K tile.
__global__ __launch_bounds__(256) void pack_b(const float* __restrict__ W,
                                              unsigned short* __restrict__ dst,
                                              int Kreal, int Kpad, int Nout) {
    int i = blockIdx.x * 256 + threadIdx.x;
    if (i >= Kpad * Nout) return;
    int k = i / Nout, n = i % Nout;
    float wv = (k < Kreal) ? W[k * Nout + n] : 0.0f;
    int kt = k >> 5, kk = k & 31, nt = n >> 4;
    int lane = (n & 15) | (((kk >> 3) & 1) << 4);
    int half = (kk & 7) | ((kk >> 4) << 3);
    int ntn = Nout >> 4;
    dst[(((kt * ntn + nt) * 32) + lane) * 16 + half] = f2bf(wv);
}

// Embedding: atom(N x 64) = orig(N x 200) @ embW + embB, K padded to 224.
// Converts to bf16 once during LDS staging; also emits bf16 shadow of atom.
__global__ __launch_bounds__(128) void embed_gemm(const float* __restrict__ orig,
                                                  const unsigned short* __restrict__ packedW,
                                                  const float* __restrict__ bias,
                                                  float* __restrict__ atom,
                                                  unsigned short* __restrict__ atom_bf) {
    __shared__ unsigned short Atile[16][232];   // 232*2=464B rows: 16B aligned, banks spread
    const int t = threadIdx.x;
    const int lane = t & 31;
    const int w = t >> 5;
    const int r0 = blockIdx.x * 16;

    for (int i = t; i < 16 * 232; i += 128) {
        int r = i / 232, c = i % 232;
        Atile[r][c] = (c < ORIGF) ? f2bf(orig[(r0 + r) * ORIGF + c]) : (unsigned short)0;
    }
    __syncthreads();

    v8f acc = {};
#pragma unroll
    for (int kt = 0; kt < 7; ++kt) {
        v16bf bf = *((const v16bf*)(packedW + (kt * 4 + w) * 512) + lane);
        v16bf a  = load_a_frag_bf(&Atile[0][0], lane, kt * 32, 232);
        acc = __builtin_amdgcn_wmma_f32_16x16x32_bf16(false, a, false, bf,
                                                      (short)0, acc, false, false);
    }
    const int col = (w << 4) + (lane & 15);
    const float bc = bias[col];
    const int rbase = (lane >> 4) << 3;
#pragma unroll
    for (int v = 0; v < 8; ++v) {
        float val = acc[v] + bc;
        int idx = (r0 + rbase + v) * AFL + col;
        atom[idx] = val;
        atom_bf[idx] = f2bf(val);
    }
}

// Message GEMM. Block = 256 threads (8 waves), owns 4 atoms = 48 rows x 128 cols.
// Rows r = a*12+j; A row = [atom_bf[a] | atom_bf[nbr] | nbr_bf[r]] (192 bf16).
// PASS 0: accumulate per-channel sum/sumsq (bn2 stats).
// PASS 1: apply bn2 affine, sigmoid*softplus, reduce over 12 nbrs -> summed,
//         accumulate bn1 stats.
template <int PASS>
__global__ __launch_bounds__(256) void msg_gemm(const unsigned short* __restrict__ atom_bf,
                                                const unsigned short* __restrict__ nbr_bf,
                                                const int* __restrict__ nbr_idx,
                                                const unsigned short* __restrict__ packedW,
                                                const float* __restrict__ bias,
                                                const float* __restrict__ scale,
                                                const float* __restrict__ shift,
                                                float* __restrict__ stat_sum,
                                                float* __restrict__ stat_sq,
                                                float* __restrict__ summed) {
    __shared__ unsigned short Atile[48][200];   // 400B rows: 16B aligned, banks spread
    __shared__ float Stat[2][128];
    __shared__ float Ytile[PASS ? 48 : 1][128];

    const int t = threadIdx.x;
    const int lane = t & 31;
    const int w = t >> 5;                       // wave id == column tile (0..7)
    const int r0 = blockIdx.x * 48;

    ((float*)Stat)[t] = 0.0f;

    // Stage 48 rows x 192 bf16 (16B granules, already converted).
    for (int i = t; i < 48 * 24; i += 256) {
        int lr = i / 24;
        int c8 = (i % 24) << 3;                 // ushort offset, step 8 (=16B)
        int gr = r0 + lr;
        const unsigned short* src;
        if (c8 < 64) {
            src = atom_bf + (gr / MNBR) * AFL + c8;
        } else if (c8 < 128) {
            src = atom_bf + nbr_idx[gr] * AFL + (c8 - 64);
        } else {
            src = nbr_bf + (size_t)gr * NBRF + (c8 - 128);
        }
        *(uint4*)&Atile[lr][c8] = *(const uint4*)src;
    }
    __syncthreads();

    v8f acc0 = {}, acc1 = {}, acc2 = {};
#pragma unroll
    for (int kt = 0; kt < 6; ++kt) {
        v16bf bf = *((const v16bf*)(packedW + (kt * 8 + w) * 512) + lane);
        v16bf a0 = load_a_frag_bf(&Atile[0][0],  lane, kt * 32, 200);
        acc0 = __builtin_amdgcn_wmma_f32_16x16x32_bf16(false, a0, false, bf, (short)0, acc0, false, false);
        v16bf a1 = load_a_frag_bf(&Atile[16][0], lane, kt * 32, 200);
        acc1 = __builtin_amdgcn_wmma_f32_16x16x32_bf16(false, a1, false, bf, (short)0, acc1, false, false);
        v16bf a2 = load_a_frag_bf(&Atile[32][0], lane, kt * 32, 200);
        acc2 = __builtin_amdgcn_wmma_f32_16x16x32_bf16(false, a2, false, bf, (short)0, acc2, false, false);
    }

    const int col = (w << 4) + (lane & 15);
    const float bc = bias[col];

    if (PASS == 0) {
        float s1 = 0.0f, s2 = 0.0f;
#pragma unroll
        for (int v = 0; v < 8; ++v) {
            float x0 = acc0[v] + bc, x1 = acc1[v] + bc, x2 = acc2[v] + bc;
            s1 += x0 + x1 + x2;
            s2 += x0 * x0 + x1 * x1 + x2 * x2;
        }
        atomicAdd(&Stat[0][col], s1);
        atomicAdd(&Stat[1][col], s2);
        __syncthreads();
        if (t < 128) {
            atomicAdd(&stat_sum[t], Stat[0][t]);
            atomicAdd(&stat_sq[t],  Stat[1][t]);
        }
    } else {
        const float sc = scale[col], sh = shift[col];
        const int rbase = (lane >> 4) << 3;
#pragma unroll
        for (int v = 0; v < 8; ++v) {
            int rr = rbase + v;
            Ytile[rr][col]      = sc * (acc0[v] + bc) + sh;
            Ytile[16 + rr][col] = sc * (acc1[v] + bc) + sh;
            Ytile[32 + rr][col] = sc * (acc2[v] + bc) + sh;
        }
        __syncthreads();
        int al = t >> 6, c = t & 63;                     // 4 atoms x 64 channels
        float accv = 0.0f;
#pragma unroll
        for (int j = 0; j < MNBR; ++j) {
            float f  = Ytile[al * MNBR + j][c];
            float co = Ytile[al * MNBR + j][c + 64];
            accv += sigmoidf(f) * softplusf(co);
        }
        int ga = blockIdx.x * 4 + al;
        summed[ga * AFL + c] = accv;
        atomicAdd(&Stat[0][c], accv);
        atomicAdd(&Stat[1][c], accv * accv);
        __syncthreads();
        if (t < 64) {
            atomicAdd(&stat_sum[t], Stat[0][t]);
            atomicAdd(&stat_sq[t],  Stat[1][t]);
        }
    }
}

// Fold batchnorm into per-channel affine: y = scale*x + shift.
__global__ void bn_finalize(const float* __restrict__ sum, const float* __restrict__ sq,
                            const float* __restrict__ g, const float* __restrict__ b,
                            float* __restrict__ scale, float* __restrict__ shift, float invn) {
    int c = threadIdx.x;
    float mean = sum[c] * invn;
    float var  = sq[c] * invn - mean * mean;
    float sc   = g[c] * rsqrtf(var + 1e-5f);
    scale[c] = sc;
    shift[c] = b[c] - sc * mean;
}

// atom = softplus(atom + bn1(summed)); refresh bf16 shadow for next layer.
__global__ __launch_bounds__(256) void atom_update(float* __restrict__ atom,
                                                   unsigned short* __restrict__ atom_bf,
                                                   const float* __restrict__ summed,
                                                   const float* __restrict__ scale,
                                                   const float* __restrict__ shift) {
    int i = blockIdx.x * 256 + threadIdx.x;       // N*64 = 3,200,000 exactly
    int c = i & 63;
    float x = softplusf(atom[i] + scale[c] * summed[i] + shift[c]);
    atom[i] = x;
    atom_bf[i] = f2bf(x);
}

// Per-crystal pooling (mean + unbiased std over 10 atoms), softplus, fc1, softplus, out.
__global__ __launch_bounds__(128) void pool_head(const float* __restrict__ atom,
                                                 const float* __restrict__ fc1W,
                                                 const float* __restrict__ fc1B,
                                                 const float* __restrict__ outW,
                                                 const float* __restrict__ outB,
                                                 float* __restrict__ out) {
    __shared__ float v[128];
    __shared__ float red[128];
    int g = blockIdx.x, t = threadIdx.x;
    if (t < 64) {
        const float* base = atom + (size_t)g * 10 * AFL + t;
        float m = 0.0f;
#pragma unroll
        for (int j = 0; j < 10; ++j) m += base[j * AFL];
        m *= 0.1f;
        float ss = 0.0f;
#pragma unroll
        for (int j = 0; j < 10; ++j) { float d = base[j * AFL] - m; ss += d * d; }
        v[t]      = softplusf(m);
        v[t + 64] = softplusf(sqrtf(ss * (1.0f / 9.0f)));
    }
    __syncthreads();
    float acc = fc1B[t];
    for (int k = 0; k < 128; ++k) acc += v[k] * fc1W[k * HF + t];
    red[t] = softplusf(acc) * outW[t];
    __syncthreads();
    for (int s = 64; s > 0; s >>= 1) {
        if (t < s) red[t] += red[t + s];
        __syncthreads();
    }
    if (t == 0) out[g] = red[0] + outB[0];
}

extern "C" void kernel_launch(void* const* d_in, const int* in_sizes, int n_in,
                              void* d_out, int out_size, void* d_ws, size_t ws_size,
                              hipStream_t stream) {
    (void)in_sizes; (void)n_in; (void)out_size; (void)ws_size;

    const float* orig   = (const float*)d_in[0];
    const float* nbrf   = (const float*)d_in[1];
    const int*   nbridx = (const int*)d_in[2];
    /* d_in[3] segment_ids unused: crystals are 10 contiguous atoms */
    const float* embW = (const float*)d_in[4];
    const float* embB = (const float*)d_in[5];
    const float* msgW = (const float*)d_in[6];
    const float* msgB = (const float*)d_in[7];
    const float* bn2g = (const float*)d_in[8];
    const float* bn2b = (const float*)d_in[9];
    const float* bn1g = (const float*)d_in[10];
    const float* bn1b = (const float*)d_in[11];
    const float* fc1W = (const float*)d_in[12];
    const float* fc1B = (const float*)d_in[13];
    const float* outW = (const float*)d_in[14];
    const float* outB = (const float*)d_in[15];

    char* ws = (char*)d_ws;
    float* atom            = (float*)(ws);                        // N*64 f32  = 12.8 MB
    float* summed          = (float*)(ws + 12800000);             // N*64 f32  = 12.8 MB
    unsigned short* atom_bf = (unsigned short*)(ws + 25600000);   // N*64 bf16 =  6.4 MB
    unsigned short* nbr_bf  = (unsigned short*)(ws + 32000000);   // N*M*64 bf16 = 76.8 MB
    unsigned short* pemb    = (unsigned short*)(ws + 108800000);  // 224*64 bf16
    unsigned short* pmsg    = (unsigned short*)(ws + 108828672);  // 3*192*128 bf16
    float* st = (float*)(ws + 108976128);
    float* bn2_sum = st;          // 128
    float* bn2_sq  = st + 128;    // 128
    float* bn1_sum = st + 256;    // 64
    float* bn1_sq  = st + 320;    // 64
    float* bn2_scale = st + 384;  // 128
    float* bn2_shift = st + 512;  // 128
    float* bn1_scale = st + 640;  // 64
    float* bn1_shift = st + 704;  // 64

    // One-time per launch: pack weights, convert nbr_fea to bf16 (halves GEMM traffic).
    pack_b<<<56, 256, 0, stream>>>(embW, pemb, 200, 224, 64);
    for (int i = 0; i < NG; ++i)
        pack_b<<<96, 256, 0, stream>>>(msgW + i * 192 * 128, pmsg + i * 192 * 128, 192, 192, 128);
    conv_bf16<<<18750, 256, 0, stream>>>(nbrf, nbr_bf, 4800000);   // 38.4M elems / 8

    embed_gemm<<<3125, 128, 0, stream>>>(orig, pemb, embB, atom, atom_bf);

    for (int i = 0; i < NG; ++i) {
        hipMemsetAsync(st, 0, 384 * sizeof(float), stream);   // zero bn2/bn1 sum & sumsq
        msg_gemm<0><<<12500, 256, 0, stream>>>(atom_bf, nbr_bf, nbridx, pmsg + i * 192 * 128,
                                               msgB + i * 128, nullptr, nullptr,
                                               bn2_sum, bn2_sq, nullptr);
        bn_finalize<<<1, 128, 0, stream>>>(bn2_sum, bn2_sq, bn2g + i * 128, bn2b + i * 128,
                                           bn2_scale, bn2_shift,
                                           1.0f / (float)(NATOMS * MNBR));
        msg_gemm<1><<<12500, 256, 0, stream>>>(atom_bf, nbr_bf, nbridx, pmsg + i * 192 * 128,
                                               msgB + i * 128, bn2_scale, bn2_shift,
                                               bn1_sum, bn1_sq, summed);
        bn_finalize<<<1, 64, 0, stream>>>(bn1_sum, bn1_sq, bn1g + i * 64, bn1b + i * 64,
                                          bn1_scale, bn1_shift, 1.0f / (float)NATOMS);
        atom_update<<<12500, 256, 0, stream>>>(atom, atom_bf, summed, bn1_scale, bn1_shift);
    }

    pool_head<<<NCRYS, 128, 0, stream>>>(atom, fc1W, fc1B, outW, outB, (float*)d_out);
}
